// AutoNER_with_RL_66984309949050
// MI455X (gfx1250) — compile-verified, matrix operand order checked by reading, not run on hardware
//
#include <hip/hip_runtime.h>
#include <hip/hip_bf16.h>
#include <math.h>

// ---- problem constants (match reference) ----
#define B_SENT   4096
#define TOK_PER  32        // T/B
#define D_DIM    512
#define NNE      32768
#define NE_LEN   8
#define E_DIM    256
#define NE_DIM   256
#define H_DIM    256
#define KDIM     768       // 3*E (im2col depth)
#define NCHUNK   24        // 768/32 K-chunks for 16x16x32 WMMA

typedef __attribute__((ext_vector_type(16))) __bf16 v16bf;
typedef __attribute__((ext_vector_type(8)))  __bf16 v8bf;
typedef __attribute__((ext_vector_type(8)))  float  v8f;

// =====================================================================
// Kernel 1: fused per-sentence attention pooling (32 tokens x 512 dims)
//   attn = h @ W_att + b;  softmax over 32 tokens;  RL_state = sum(h*e)/sum(e)
// One block per sentence, 256 threads (8 waves). Bandwidth-bound on h.
// =====================================================================
__global__ __launch_bounds__(256) void attn_pool_kernel(
    const float* __restrict__ h, const float* __restrict__ W_att,
    const float* __restrict__ b_att, float* __restrict__ RL_state)
{
  __shared__ __align__(16) float watt[512];
  __shared__ float attnS[TOK_PER];
  __shared__ float evS[TOK_PER];
  __shared__ float denomS;

  const int tid = threadIdx.x;
  const int b   = blockIdx.x;
  const float* hb = h + (size_t)b * TOK_PER * D_DIM;

  watt[tid]       = W_att[tid];
  watt[tid + 256] = W_att[tid + 256];
  __syncthreads();

  // phase A: attention scores. 8 lanes per token, each lane dots 64 contiguous dims.
  {
    const int tok = tid >> 3, lg = tid & 7;
    const float4* hr4 = (const float4*)(hb + tok * D_DIM + lg * 64);
    const float*  wv  = &watt[lg * 64];
    float p = 0.f;
    #pragma unroll
    for (int u = 0; u < 16; ++u) {
      float4 hv = hr4[u];
      p += hv.x * wv[u*4+0] + hv.y * wv[u*4+1] + hv.z * wv[u*4+2] + hv.w * wv[u*4+3];
    }
    p += __shfl_xor(p, 4);
    p += __shfl_xor(p, 2);
    p += __shfl_xor(p, 1);
    if (lg == 0) attnS[tok] = p + b_att[0];
  }
  __syncthreads();

  // phase B: 32-wide segment softmax in one wave (wave32 == segment length)
  if (tid < 32) {
    float a = attnS[tid];
    float m = a;
    #pragma unroll
    for (int off = 16; off >= 1; off >>= 1) m = fmaxf(m, __shfl_xor(m, off));
    float e = __expf(a - m);
    float s = e;
    #pragma unroll
    for (int off = 16; off >= 1; off >>= 1) s += __shfl_xor(s, off);
    evS[tid] = e;
    if (tid == 0) denomS = s;
  }
  __syncthreads();

  // phase C: weighted sum (second pass over h; L2-resident). 2 dims per thread.
  const float inv = 1.0f / denomS;
  #pragma unroll
  for (int q = 0; q < 2; ++q) {
    const int d = tid + q * 256;
    float acc = 0.f;
    #pragma unroll 8
    for (int t = 0; t < TOK_PER; ++t) acc += hb[t * D_DIM + d] * evS[t];
    RL_state[(size_t)b * D_DIM + d] = acc * inv;
  }
}

// =====================================================================
// Prep kernels: transpose + bf16-convert weights for contiguous B-fragment
// loads (Bt[o][kw*256+i] for conv, W1t[o][j] for the gate GEMM).
// =====================================================================
__global__ __launch_bounds__(256) void build_bt_kernel(
    const float* __restrict__ conv_w, __bf16* __restrict__ Bt)
{
  const int idx = blockIdx.x * 256 + threadIdx.x;   // 256*768 elements
  const int o  = idx / KDIM;
  const int kk = idx % KDIM;
  const int kw = kk >> 8, i = kk & 255;
  Bt[idx] = (__bf16)conv_w[(o * E_DIM + i) * 3 + kw];   // conv_w [O][I][3]
}

__global__ __launch_bounds__(256) void build_w1t_kernel(
    const float* __restrict__ W1, __bf16* __restrict__ W1t)
{
  const int idx = blockIdx.x * 256 + threadIdx.x;   // 256*768 elements
  const int o = idx / KDIM;
  const int j = idx % KDIM;
  W1t[idx] = (__bf16)W1[j * H_DIM + o];             // W1 [768][256]
}

// =====================================================================
// Kernel 2: NE conv-as-GEMM (bf16 WMMA) + fused maxpool + fused segment
// mean.  One workgroup per SENTENCE: 8 entities -> M=64 im2col rows.
//   Wave w owns N cols [w*64, w*64+64): 4 N-tiles x 4 M-tiles = 16 acc
//   tiles (128 VGPRs).  K = 768 in 24 chunks.  B fragments use one-deep
//   register prefetch so each wait covers loads issued 4 WMMAs earlier.
// Epilogue: per-lane 8-way max (the 8 conv positions of one entity live in
// one lane's accumulator), sum of 4 entities per lane, shfl_xor(16) to add
// the other 4 -> NE_state written directly (no ne_feat round-trip).
// =====================================================================
#define EPAD 264   // 256 + 8 bf16 pad -> row stride 528B, bank shift 4 dwords

__global__ __launch_bounds__(128) void ne_conv_gemm_kernel(
    const float* __restrict__ W_emb, const int* __restrict__ NE_ids,
    const float* __restrict__ conv_b, const __bf16* __restrict__ Bt,
    float* __restrict__ NE_state)
{
  // staged embeddings: [8 ents][10 rows][EPAD] bf16; rows 0/9 are zero halo
  __shared__ __align__(16) __bf16 embs[8 * 10 * EPAD];
  const int tid = threadIdx.x;
  const int wg  = blockIdx.x;           // sentence; entities 8*wg .. 8*wg+7

  // zero the halo rows (l_src = -1 and l_src = 8) for all 8 entities
  for (int z = tid; z < 16 * EPAD; z += 128) {
    const int which = z / EPAD;          // 0..15
    const int col   = z % EPAD;
    const int e     = which >> 1;
    const int row   = (which & 1) ? 9 : 0;
    embs[(e * 10 + row) * EPAD + col] = (__bf16)0.0f;
  }
  // gather: 64 (ent,l) rows, 2 threads/row, 128 floats each (f32 -> bf16)
  {
    const int r = tid >> 1, sub = tid & 1;
    const int e = r >> 3, l = r & 7;
    const int id = NE_ids[(8 * wg + e) * NE_LEN + l];
    const float4* src = (const float4*)(W_emb + (size_t)id * E_DIM) + sub * 32;
    __bf16* dst = &embs[(e * 10 + l + 1) * EPAD + sub * 128];
    #pragma unroll
    for (int u = 0; u < 32; ++u) {
      float4 v = src[u];
      dst[u*4+0] = (__bf16)v.x; dst[u*4+1] = (__bf16)v.y;
      dst[u*4+2] = (__bf16)v.z; dst[u*4+3] = (__bf16)v.w;
    }
  }
  __syncthreads();

  const int lane = tid & 31;
  const int wave = tid >> 5;
  const int hi   = lane >> 4;           // K-subset selector (ISA A/B layout)
  const int mrow = lane & 15;           // A row / B column within tile
  const int n0   = wave * 64;

  v8f acc[4][4] = {};                   // [nt][mt]
  for (int c = 0; c < NCHUNK; ++c) {
    // A fragments for all 4 M-tiles (8 x ds_load_b128, hoisted together)
    v16bf a[4];
    #pragma unroll
    for (int mt = 0; mt < 4; ++mt) {
      #pragma unroll
      for (int hh = 0; hh < 2; ++hh) {
        const int kk = c * 32 + hi * 8 + hh * 16;
        const int kw = kk >> 8;
        const int i0 = kk & 255;
        const int M  = mt * 16 + mrow;
        const int e  = M >> 3;
        const int l  = M & 7;
        const v8bf av = *(const v8bf*)&embs[(e * 10 + l + kw) * EPAD + i0];
        #pragma unroll
        for (int j = 0; j < 8; ++j) a[mt][hh * 8 + j] = av[j];
      }
    }
    // B fragments: one-deep register prefetch across the 4 N-tiles
    const __bf16* bbase = Bt + c * 32 + hi * 8;
    v8bf blo = *(const v8bf*)(bbase + (size_t)(n0 + mrow) * KDIM);
    v8bf bhc = *(const v8bf*)(bbase + (size_t)(n0 + mrow) * KDIM + 16);
    #pragma unroll
    for (int nt = 0; nt < 4; ++nt) {
      v8bf nlo = {}, nhc = {};
      if (nt < 3) {
        const __bf16* bp = bbase + (size_t)(n0 + (nt + 1) * 16 + mrow) * KDIM;
        nlo = *(const v8bf*)bp;
        nhc = *(const v8bf*)(bp + 16);
      }
      v16bf bf;
      #pragma unroll
      for (int j = 0; j < 8; ++j) { bf[j] = blo[j]; bf[8 + j] = bhc[j]; }
      #pragma unroll
      for (int mt = 0; mt < 4; ++mt)
        acc[nt][mt] = __builtin_amdgcn_wmma_f32_16x16x32_bf16(
            false, a[mt], false, bf, (short)0, acc[nt][mt], false, false);
      if (nt < 3) { blo = nlo; bhc = nhc; }
    }
  }

  // epilogue: maxpool over 8 conv positions (register-local per lane) then
  // mean over the sentence's 8 entities (4 per lane + shfl_xor across hi).
  #pragma unroll
  for (int nt = 0; nt < 4; ++nt) {
    const int ncol = n0 + nt * 16 + mrow;
    const float cb = conv_b[ncol];
    float s = 0.f;
    #pragma unroll
    for (int mt = 0; mt < 4; ++mt) {     // entity = mt*2 + hi
      v8f y = acc[nt][mt];
      float m = y[0];
      #pragma unroll
      for (int j = 1; j < 8; ++j) m = fmaxf(m, y[j]);
      s += m + cb;
    }
    s += __shfl_xor(s, 16);              // add the other 4 entities
    if (hi == 0) NE_state[(size_t)wg * NE_DIM + ncol] = s * 0.125f;
  }
}

// =====================================================================
// Kernel 3: RL gate. [16 sentences x 768] @ W1 via WMMA, then fold
// (+b1) @ W2 + b2 and sigmoid into the epilogue with wave32 shuffles.
// =====================================================================
#define SPAD 776   // 768 + 8 bf16 pad

__global__ __launch_bounds__(128) void gate_kernel(
    const float* __restrict__ RL_state, const float* __restrict__ NE_state,
    const __bf16* __restrict__ W1t, const float* __restrict__ b1,
    const float* __restrict__ W2, const float* __restrict__ b2,
    float* __restrict__ out)
{
  __shared__ __align__(16) __bf16 st[16 * SPAD];
  __shared__ float psum[4][16];
  const int tid = threadIdx.x, wg = blockIdx.x;

  // stage concat state [RL | NE] for 16 sentences, f32 -> bf16
  for (int idx = tid; idx < 16 * KDIM; idx += 128) {
    const int m = idx / KDIM, j = idx % KDIM;
    const int s = wg * 16 + m;
    const float v = (j < D_DIM) ? RL_state[(size_t)s * D_DIM + j]
                                : NE_state[(size_t)s * NE_DIM + (j - D_DIM)];
    st[m * SPAD + j] = (__bf16)v;
  }
  __syncthreads();

  const int lane = tid & 31, wave = tid >> 5;
  const int hi = lane >> 4, mrow = lane & 15;
  const int n0 = wave * 64;

  v8f acc[4] = {};
  for (int c = 0; c < NCHUNK; ++c) {
    v16bf a;
    #pragma unroll
    for (int hh = 0; hh < 2; ++hh) {
      const int kk = c * 32 + hi * 8 + hh * 16;
      const v8bf av = *(const v8bf*)&st[mrow * SPAD + kk];
      #pragma unroll
      for (int j = 0; j < 8; ++j) a[hh * 8 + j] = av[j];
    }
    const __bf16* bbase = W1t + c * 32 + hi * 8;
    v8bf blo = *(const v8bf*)(bbase + (size_t)(n0 + mrow) * KDIM);
    v8bf bhc = *(const v8bf*)(bbase + (size_t)(n0 + mrow) * KDIM + 16);
    #pragma unroll
    for (int nt = 0; nt < 4; ++nt) {
      v8bf nlo = {}, nhc = {};
      if (nt < 3) {
        const __bf16* bp = bbase + (size_t)(n0 + (nt + 1) * 16 + mrow) * KDIM;
        nlo = *(const v8bf*)bp;
        nhc = *(const v8bf*)(bp + 16);
      }
      v16bf bf;
      #pragma unroll
      for (int j = 0; j < 8; ++j) { bf[j] = blo[j]; bf[8 + j] = bhc[j]; }
      acc[nt] = __builtin_amdgcn_wmma_f32_16x16x32_bf16(
          false, a, false, bf, (short)0, acc[nt], false, false);
      if (nt < 3) { blo = nlo; bhc = nhc; }
    }
  }

  // epilogue: per lane, elements r hold hidden[m=r(+8*hi)][n=ncol].
  // fold (hidden + b1) * W2, reduce over the 16 n-lanes, then over 4 waves.
  float part[8];
  #pragma unroll
  for (int r = 0; r < 8; ++r) part[r] = 0.f;
  #pragma unroll
  for (int nt = 0; nt < 4; ++nt) {
    const int ncol = n0 + nt * 16 + mrow;
    const float w2v = W2[ncol];
    const float b1v = b1[ncol];
    #pragma unroll
    for (int r = 0; r < 8; ++r) part[r] += (acc[nt][r] + b1v) * w2v;
  }
  #pragma unroll
  for (int off = 8; off >= 1; off >>= 1) {
    #pragma unroll
    for (int r = 0; r < 8; ++r) part[r] += __shfl_xor(part[r], off);
  }
  if (mrow == 0) {
    #pragma unroll
    for (int r = 0; r < 8; ++r) psum[wave][hi * 8 + r] = part[r];
  }
  __syncthreads();
  if (tid < 16) {
    const float s = psum[0][tid] + psum[1][tid] + psum[2][tid] + psum[3][tid] + b2[0];
    out[wg * 16 + tid] = 1.f / (1.f + __expf(-s));
  }
}

// =====================================================================
// Launch: inputs in setup_inputs() order.
// ws layout: RL_state(8MB) | NE_state(4MB) | Bt(384KB) | W1t(384KB)
// =====================================================================
extern "C" void kernel_launch(void* const* d_in, const int* in_sizes, int n_in,
                              void* d_out, int out_size, void* d_ws, size_t ws_size,
                              hipStream_t stream) {
  const float* h      = (const float*)d_in[0];
  const float* W_emb  = (const float*)d_in[1];
  const float* W_att  = (const float*)d_in[2];
  const float* b_att  = (const float*)d_in[3];
  const float* conv_w = (const float*)d_in[4];
  const float* conv_b = (const float*)d_in[5];
  const float* W1     = (const float*)d_in[6];
  const float* b1     = (const float*)d_in[7];
  const float* W2     = (const float*)d_in[8];
  const float* b2     = (const float*)d_in[9];
  const int*   NE_ids = (const int*)d_in[10];
  float* out = (float*)d_out;

  char* ws = (char*)d_ws;
  float*  RL_state = (float*)(ws);                              //  8 MB
  float*  NE_state = (float*)(ws + 8388608);                    //  4 MB
  __bf16* Bt       = (__bf16*)(ws + 12582912);                  // 384 KB
  __bf16* W1t      = (__bf16*)(ws + 12976128);                  // 384 KB

  attn_pool_kernel<<<B_SENT, 256, 0, stream>>>(h, W_att, b_att, RL_state);
  build_bt_kernel<<<768, 256, 0, stream>>>(conv_w, Bt);
  build_w1t_kernel<<<768, 256, 0, stream>>>(W1, W1t);
  ne_conv_gemm_kernel<<<B_SENT, 128, 0, stream>>>(W_emb, NE_ids, conv_b, Bt, NE_state);
  gate_kernel<<<B_SENT / 16, 128, 0, stream>>>(RL_state, NE_state, W1t, b1, W2, b2, out);
}
